// NetworkD_41137196761155
// MI455X (gfx1250) — matrix operand order repaired and checked by reading, not compiled, not run on hardware
//
#include <hip/hip_runtime.h>

// ---------------------------------------------------------------------------
// Types for CDNA5 WMMA (wave32, gfx1250)
// ---------------------------------------------------------------------------
typedef __bf16 bf16;
typedef __attribute__((ext_vector_type(16))) __bf16 v16bf;
typedef __attribute__((ext_vector_type(8)))  float  v8f;
typedef __attribute__((ext_vector_type(4)))  unsigned int u32x4;

union FragBF {
    u32x4 u[2];   // two 16B chunks
    v16bf v;      // 16 bf16 per lane = 8 VGPRs
};

__device__ __forceinline__ bf16 to_bf16(float f) { return (bf16)f; }

// Async global->LDS (CDNA5 GLOBAL_LOAD_ASYNC_TO_LDS_B128, ASYNCcnt tracked)
#if defined(__AMDGCN__) && __has_builtin(__builtin_amdgcn_global_load_async_to_lds_b128) && __has_builtin(__builtin_amdgcn_s_wait_asynccnt)
#define USE_ASYNC_LDS 1
#else
#define USE_ASYNC_LDS 0
#endif

#if USE_ASYNC_LDS
// builtin signature (from hipcc diagnostic): (int4 AS1*, int4 AS3*, imm offset, imm cpol)
typedef int v4i_ __attribute__((vector_size(16)));
typedef __attribute__((address_space(1))) v4i_ as1_v4i;
typedef __attribute__((address_space(3))) v4i_ as3_v4i;

__device__ __forceinline__ void async_copy16(void* lds, const void* gsrc) {
    __builtin_amdgcn_global_load_async_to_lds_b128(
        (as1_v4i*)(v4i_*)gsrc,
        (as3_v4i*)lds,
        0, 0);
}
#endif

#if defined(__AMDGCN__) && __has_builtin(__builtin_amdgcn_sched_group_barrier)
#define HAVE_SCHED_GROUP 1
#else
#define HAVE_SCHED_GROUP 0
#endif

// ---------------------------------------------------------------------------
// TT weight materialization:
//   Q[(iO*m+im)][(jO*n+jn)][q] = sum_p P[iO][jO][p] * G[p][im][jn][q]
// P layout [I][J][r0], G layout [r0][m][n][r1], Q layout [I*m][J*n][r1].
// ---------------------------------------------------------------------------
__global__ void tt_expand(const float* __restrict__ P, const float* __restrict__ G,
                          float* __restrict__ Q,
                          int I, int J, int r0, int m, int n, int r1, int total) {
    int idx = blockIdx.x * blockDim.x + threadIdx.x;
    if (idx >= total) return;
    int t  = idx;
    int q  = t % r1; t /= r1;
    int jn = t % n;  t /= n;
    int jO = t % J;  t /= J;
    int im = t % m;
    int iO = t / m;
    (void)I;
    float s = 0.f;
    const float* prow = P + ((size_t)iO * J + jO) * r0;
    for (int p = 0; p < r0; ++p)
        s += prow[p] * G[(((size_t)p * m + im) * n + jn) * r1 + q];
    Q[idx] = s;
}

// Final step: r4 == 1, so W[(iL*m4+i4)][(jL*n4+j4)] = P4[iL][jL] * G4[i4][j4].
// Emits bf16, row-major [K][N].
__global__ void tt_final_w(const float* __restrict__ P4, const float* __restrict__ G4,
                           bf16* __restrict__ W,
                           int JL, int m4, int n4, int N, int total) {
    int idx = blockIdx.x * blockDim.x + threadIdx.x;
    if (idx >= total) return;
    int j = idx % N;
    int i = idx / N;
    int iL = i / m4, i4 = i % m4;
    int jL = j / n4, j4 = j % n4;
    W[idx] = to_bf16(P4[(size_t)iL * JL + jL] * G4[i4 * n4 + j4]);
}

__global__ void f32_to_bf16_kernel(const float* __restrict__ X, bf16* __restrict__ Y, int n) {
    int i = blockIdx.x * blockDim.x + threadIdx.x;
    if (i < n) Y[i] = to_bf16(X[i]);
}

// ---------------------------------------------------------------------------
// Dense GEMM, C = relu(A * W + bias):  A[M,K] bf16 row-major, W[K,N] bf16
// row-major, C[M,N] bf16.  128x128 block tile, BK=64, 256 threads = 8 waves,
// each wave computes 32x64 via 16 x v_wmma_f32_16x16x32_bf16 per K-tile.
// Double-buffered LDS; async DMA staging with a 2-tile pipeline; WMMA/DS
// software pipelining pinned with sched_group_barrier.
// Requires: M % 128 == 0, N % 128 == 0, K % 64 == 0.
// ---------------------------------------------------------------------------
#define BM 128
#define BN 128
#define BK 64

struct SmemBufs { bf16 a[2][BM * BK]; bf16 b[2][BK * BN]; };  // 2 x (16KB+16KB)
union  Smem     { SmemBufs s; bf16 c[BM * BN]; };             // 64KB total

__global__ __launch_bounds__(256)
void gemm_bf16_relu(const bf16* __restrict__ A, const bf16* __restrict__ W,
                    const float* __restrict__ bias, bf16* __restrict__ C,
                    int N, int K) {
    __shared__ Smem sm;
    const int tid  = threadIdx.x;
    const int lane = tid & 31;
    const int wave = tid >> 5;
    const int wm   = wave & 3;   // 4 row-slabs of 32
    const int wn   = wave >> 2;  // 2 col-slabs of 64
    const int bn0  = blockIdx.x * BN;
    const int bm0  = blockIdx.y * BM;

    // ---- staging geometry: 1024 16B-chunks per tile, 4 A + 4 B per thread
    const int arow = tid >> 3;            // 0..31  (+0/32/64/96)
    const int akc  = (tid & 7) * 8;       // 0..56
    const int brow = tid >> 4;            // 0..15  (+0/16/32/48)
    const int bnc  = (tid & 15) * 8;      // 0..120

    const bf16* pA = A + (size_t)(bm0 + arow) * K + akc;
    const bf16* pB = W + (size_t)brow * N + bn0 + bnc;
    const size_t aRow = (size_t)32 * K;   // stride between a thread's A chunks
    const size_t bRow = (size_t)16 * N;   // stride between a thread's B chunks
    const int aoff = arow * BK + akc;     // LDS element offsets
    const int boff = brow * BN + bnc;

    v8f acc[2][4] = {};
    const int nK = K / BK;

    auto issueTile = [&](int buf) {
#if USE_ASYNC_LDS
#pragma unroll
        for (int i = 0; i < 4; ++i)
            async_copy16(&sm.s.a[buf][aoff + i * 32 * BK], pA + (size_t)i * aRow);
#pragma unroll
        for (int i = 0; i < 4; ++i)
            async_copy16(&sm.s.b[buf][boff + i * 16 * BN], pB + (size_t)i * bRow);
#else
        u32x4 ra[4], rb[4];
#pragma unroll
        for (int i = 0; i < 4; ++i) ra[i] = *(const u32x4*)(pA + (size_t)i * aRow);
#pragma unroll
        for (int i = 0; i < 4; ++i) rb[i] = *(const u32x4*)(pB + (size_t)i * bRow);
#pragma unroll
        for (int i = 0; i < 4; ++i) *(u32x4*)&sm.s.a[buf][aoff + i * 32 * BK] = ra[i];
#pragma unroll
        for (int i = 0; i < 4; ++i) *(u32x4*)&sm.s.b[buf][boff + i * 16 * BN] = rb[i];
#endif
        pA += BK;               // advance one K-tile
        pB += (size_t)BK * N;
    };

    auto computeTile = [&](int buf) {
        const int r0 = wm * 32 + (lane & 15);
        const int kh = (lane >> 4) * 8;
#pragma unroll
        for (int ks = 0; ks < 2; ++ks) {
            FragBF af[2], fb[4];
#pragma unroll
            for (int tm = 0; tm < 2; ++tm) {
                int r = r0 + tm * 16;
                af[tm].u[0] = *(const u32x4*)&sm.s.a[buf][r * BK + ks * 32 + kh];
                af[tm].u[1] = *(const u32x4*)&sm.s.a[buf][r * BK + ks * 32 + 16 + kh];
            }
#pragma unroll
            for (int tn = 0; tn < 4; ++tn) {
                int bc = wn * 64 + tn * 16;
#pragma unroll
                for (int s = 0; s < 8; ++s) {
                    fb[tn].v[s]     = sm.s.b[buf][(ks * 32 + kh + s) * BN + bc + (lane & 15)];
                    fb[tn].v[8 + s] = sm.s.b[buf][(ks * 32 + 16 + kh + s) * BN + bc + (lane & 15)];
                }
            }
#pragma unroll
            for (int tn = 0; tn < 4; ++tn)
#pragma unroll
                for (int tm = 0; tm < 2; ++tm) {
#if defined(__gfx1250__)
                    acc[tm][tn] = __builtin_amdgcn_wmma_f32_16x16x32_bf16(
                        false, af[tm].v, false, fb[tn].v, (short)0, acc[tm][tn], false, false);
#else
                    acc[tm][tn][0] += (float)fb[tn].v[0] * (float)af[tm].v[0]; // host stub
#endif
                }
#if HAVE_SCHED_GROUP
            // Pin a software-pipelined schedule per 32-K substep:
            //   [A frags + B0 : 6 DS reads] [B1: 2 DS] [wmma pair0]
            //   [B2: 2 DS] [wmma pair1] [B3: 2 DS] [wmma pair2] [wmma pair3]
            // -> every wmma pair's LDS operands were issued one group earlier,
            //    so dscnt waits are staged instead of 0 (latency hidden).
            __builtin_amdgcn_sched_group_barrier(0x100, 6, 0);  // DS read x6
            __builtin_amdgcn_sched_group_barrier(0x100, 2, 0);  // DS read x2 (B1)
            __builtin_amdgcn_sched_group_barrier(0x008, 2, 0);  // WMMA pair 0
            __builtin_amdgcn_sched_group_barrier(0x100, 2, 0);  // DS read x2 (B2)
            __builtin_amdgcn_sched_group_barrier(0x008, 2, 0);  // WMMA pair 1
            __builtin_amdgcn_sched_group_barrier(0x100, 2, 0);  // DS read x2 (B3)
            __builtin_amdgcn_sched_group_barrier(0x008, 2, 0);  // WMMA pair 2
            __builtin_amdgcn_sched_group_barrier(0x008, 2, 0);  // WMMA pair 3
#endif
        }
    };

    // ---- 2-tile-deep pipeline
    issueTile(0);
    if (nK > 1) issueTile(1);
    for (int kt = 0; kt < nK; ++kt) {
        const int cur = kt & 1;
#if USE_ASYNC_LDS
        if (kt + 1 < nK) __builtin_amdgcn_s_wait_asynccnt(8);  // only next tile in flight
        else             __builtin_amdgcn_s_wait_asynccnt(0);
#endif
        __syncthreads();
        computeTile(cur);
        __syncthreads();
        if (kt + 2 < nK) issueTile(cur);
    }

    // ---- epilogue: bias + ReLU, transpose through LDS, b128 stores
    {
        const int chi = lane >> 4;    // M += 8 for upper half-lanes
        const int cn  = lane & 15;    // N = lane&15
#pragma unroll
        for (int tm = 0; tm < 2; ++tm)
#pragma unroll
            for (int tn = 0; tn < 4; ++tn) {
                int colL = wn * 64 + tn * 16 + cn;
                float bv = bias[bn0 + colL];
#pragma unroll
                for (int r = 0; r < 8; ++r) {
                    int rowL = wm * 32 + tm * 16 + chi * 8 + r;
                    float v = acc[tm][tn][r] + bv;
                    v = v > 0.f ? v : 0.f;
                    sm.c[rowL * BN + colL] = to_bf16(v);
                }
            }
    }
    __syncthreads();
#pragma unroll
    for (int i = 0; i < 8; ++i) {
        int c = tid + i * 256;
        int r = c >> 4, nc = (c & 15) * 8;
        *(u32x4*)&C[(size_t)(bm0 + r) * N + bn0 + nc] = *(const u32x4*)&sm.c[r * BN + nc];
    }
}

// ---------------------------------------------------------------------------
// Head: logits = A[8192,2048] * W[2048,10] + bias, then row softmax (f32 out).
// One wave per row, W cached in LDS.
// ---------------------------------------------------------------------------
__global__ __launch_bounds__(256)
void head_softmax(const bf16* __restrict__ A, const bf16* __restrict__ W,
                  const float* __restrict__ bias, float* __restrict__ out,
                  int K, int NOUT) {
    __shared__ bf16 Ws[2048 * 10];
    for (int i = threadIdx.x; i < K * NOUT; i += 256) Ws[i] = W[i];
    __syncthreads();
    const int lane = threadIdx.x & 31;
    const int wave = threadIdx.x >> 5;
    const int row  = blockIdx.x * 8 + wave;

    float acc[10];
#pragma unroll
    for (int j = 0; j < 10; ++j) acc[j] = 0.f;
    for (int k = lane; k < K; k += 32) {
        float a = (float)A[(size_t)row * K + k];
#pragma unroll
        for (int j = 0; j < 10; ++j) acc[j] += a * (float)Ws[k * 10 + j];
    }
#pragma unroll
    for (int j = 0; j < 10; ++j)
        for (int off = 16; off; off >>= 1)
            acc[j] += __shfl_xor(acc[j], off, 32);

    if (lane == 0) {
        float mx = -1e30f;
#pragma unroll
        for (int j = 0; j < 10; ++j) { acc[j] += bias[j]; mx = fmaxf(mx, acc[j]); }
        float se = 0.f;
#pragma unroll
        for (int j = 0; j < 10; ++j) { acc[j] = expf(acc[j] - mx); se += acc[j]; }
        float inv = 1.f / se;
#pragma unroll
        for (int j = 0; j < 10; ++j) out[row * 10 + j] = acc[j] * inv;
    }
}

// ---------------------------------------------------------------------------
// Orchestration
// ---------------------------------------------------------------------------
extern "C" void kernel_launch(void* const* d_in, const int* in_sizes, int n_in,
                              void* d_out, int out_size, void* d_ws, size_t ws_size,
                              hipStream_t stream) {
    (void)in_sizes; (void)n_in; (void)out_size; (void)ws_size;
    static const int IM[5][5] = {{7,4,4,4,7},{8,8,8,8,4},{8,8,8,4,4},{8,8,4,4,4},{8,4,4,4,4}};
    static const int OM[5][5] = {{8,8,8,8,4},{8,8,8,4,4},{8,8,4,4,4},{8,4,4,4,4},{10,1,1,1,1}};
    const int B = 8192;

    // bump allocator on workspace (256B aligned)
    char* p = (char*)d_ws;
    auto take = [&](size_t bytes) -> char* {
        char* r = p;
        p += (bytes + 255) & ~(size_t)255;
        return r;
    };
    bf16* act[5];
    act[0] = (bf16*)take((size_t)B * 3136  * 2);
    act[1] = (bf16*)take((size_t)B * 16384 * 2);
    act[2] = (bf16*)take((size_t)B * 8192  * 2);
    act[3] = (bf16*)take((size_t)B * 4096  * 2);
    act[4] = (bf16*)take((size_t)B * 2048  * 2);
    bf16*  Wbuf = (bf16*)take((size_t)16384 * 8192 * 2);   // reused each layer
    float* P2   = (float*)take((size_t)64  * 64  * 7 * 4);
    float* P3   = (float*)take((size_t)512 * 512 * 7 * 4);
    float* P4   = (float*)take((size_t)4096 * 2048 * 4);

    // x (f32) -> bf16
    {
        int n = B * 3136;
        f32_to_bf16_kernel<<<(n + 255) / 256, 256, 0, stream>>>(
            (const float*)d_in[0], act[0], n);
    }

    for (int l = 0; l < 5; ++l) {
        const float* G[5];
        for (int c = 0; c < 5; ++c) G[c] = (const float*)d_in[1 + l * 6 + c];
        const float* bias = (const float*)d_in[1 + l * 6 + 5];
        const int K = IM[l][0]*IM[l][1]*IM[l][2]*IM[l][3]*IM[l][4];
        const int N = OM[l][0]*OM[l][1]*OM[l][2]*OM[l][3]*OM[l][4];

        // expand G0*G1 -> P2   (G0's layout [m0][n0][r1] is already "P1")
        {
            int I = IM[l][0], J = OM[l][0];
            int tot = I * IM[l][1] * J * OM[l][1] * 7;
            tt_expand<<<(tot + 255) / 256, 256, 0, stream>>>(
                G[0], G[1], P2, I, J, 7, IM[l][1], OM[l][1], 7, tot);
        }
        // P2*G2 -> P3
        {
            int I = IM[l][0]*IM[l][1], J = OM[l][0]*OM[l][1];
            int tot = I * IM[l][2] * J * OM[l][2] * 7;
            tt_expand<<<(tot + 255) / 256, 256, 0, stream>>>(
                P2, G[2], P3, I, J, 7, IM[l][2], OM[l][2], 7, tot);
        }
        // P3*G3 -> P4   (r4 == 1)
        {
            int I = IM[l][0]*IM[l][1]*IM[l][2], J = OM[l][0]*OM[l][1]*OM[l][2];
            int tot = I * IM[l][3] * J * OM[l][3];
            tt_expand<<<(tot + 255) / 256, 256, 0, stream>>>(
                P3, G[3], P4, I, J, 7, IM[l][3], OM[l][3], 1, tot);
        }
        // P4 kron G4 -> W (bf16, [K][N])
        {
            int JL = OM[l][0]*OM[l][1]*OM[l][2]*OM[l][3];
            int tot = K * N;
            tt_final_w<<<(tot + 255) / 256, 256, 0, stream>>>(
                P4, G[4], Wbuf, JL, IM[l][4], OM[l][4], N, tot);
        }

        if (l < 4) {
            dim3 grid(N / BN, B / BM);
            gemm_bf16_relu<<<grid, 256, 0, stream>>>(
                act[l], Wbuf, bias, act[l + 1], N, K);
        } else {
            head_softmax<<<B / 8, 256, 0, stream>>>(
                act[4], Wbuf, bias, (float*)d_out, K, N);
        }
    }
}
